// TimeStepAttention_833223655918
// MI455X (gfx1250) — matrix operand order repaired and checked by reading, not compiled, not run on hardware
//
#include <hip/hip_runtime.h>
#include <hip/hip_bf16.h>

// ---------------------------------------------------------------------------
// Problem constants (match reference): S=1024, B=8, E=1024, H=16, D=64
// ---------------------------------------------------------------------------
#define S_DIM 1024
#define B_DIM 8
#define E_DIM 1024
#define H_DIM 16
#define D_DIM 64
#define THREE_E 3072
#define M_DIM (S_DIM * B_DIM)     // 8192 rows of x / qkv / ctx

typedef __attribute__((ext_vector_type(16))) __bf16 v16bf;
typedef __attribute__((ext_vector_type(8)))  float  v8f;

// ---------------------------------------------------------------------------
// helpers
// ---------------------------------------------------------------------------
__device__ __forceinline__ unsigned short bf16_bits(float f) {
  unsigned u = __builtin_bit_cast(unsigned, f);
  u += 0x7FFFu + ((u >> 16) & 1u);            // round-to-nearest-even
  return (unsigned short)(u >> 16);
}

__device__ __forceinline__ __bf16 to_bf16(float f) {
  unsigned short h = bf16_bits(f);
  return __builtin_bit_cast(__bf16, h);
}

__device__ __forceinline__ v8f wmma_bf16(v16bf a, v16bf b, v8f c) {
  return __builtin_amdgcn_wmma_f32_16x16x32_bf16(
      /*neg_a=*/false, a, /*neg_b=*/false, b,
      /*c_mod=*/(short)0, c, /*reuse_a=*/false, /*reuse_b=*/false);
}

// A-matrix 16x32 bf16 fragment (contiguous k order, same as the B frags): lane = M row (mod 16);
// lane group 0 (lanes 0-15) holds K = k0+0..15; group 1 holds k0+16..31.
__device__ __forceinline__ v16bf frag_a_bf16(const __bf16* base, int ld, int row0, int k0) {
  int lane = threadIdx.x & 31;
  const __bf16* p = base + (size_t)(row0 + (lane & 15)) * ld + k0 + ((lane >> 4) << 4);
  v16bf f;
#pragma unroll
  for (int e = 0; e < 8; ++e) f[e] = p[e];        // 16B contiguous -> b128
#pragma unroll
  for (int e = 0; e < 8; ++e) f[e + 8] = p[8 + e]; // 16B contiguous -> b128
  return f;
}

// Same A layout but from f32 source (LDS attention probs), convert on read.
__device__ __forceinline__ v16bf frag_a_f32(const float* base, int ld, int row0, int k0) {
  int lane = threadIdx.x & 31;
  const float* p = base + (size_t)(row0 + (lane & 15)) * ld + k0 + ((lane >> 4) << 4);
  v16bf f;
#pragma unroll
  for (int e = 0; e < 8; ++e) f[e] = to_bf16(p[e]);
#pragma unroll
  for (int e = 0; e < 8; ++e) f[e + 8] = to_bf16(p[8 + e]);
  return f;
}

// B-matrix 32x16 bf16 fragment: lane = N column (mod 16);
// lane group 0 holds K = k0+0..15 linearly, group 1 holds k0+16..31.
// "NT" source: B[n][k] rows with K contiguous.
__device__ __forceinline__ v16bf frag_b_nt_bf16(const __bf16* base, int ld, int n0, int k0) {
  int lane = threadIdx.x & 31;
  const __bf16* p = base + (size_t)(n0 + (lane & 15)) * ld + k0 + ((lane >> 4) << 4);
  v16bf f;
#pragma unroll
  for (int e = 0; e < 16; ++e) f[e] = p[e];       // 32B contiguous -> 2x b128
  return f;
}

// "NN" source: B[k][n] with N contiguous, K strided by ld (V-matrix).
__device__ __forceinline__ v16bf frag_b_nn_bf16(const __bf16* base, int ld, int k0, int n0) {
  int lane = threadIdx.x & 31;
  const __bf16* p = base + (size_t)(k0 + ((lane >> 4) << 4)) * ld + n0 + (lane & 15);
  v16bf f;
#pragma unroll
  for (int e = 0; e < 16; ++e) f[e] = p[(size_t)e * ld];
  return f;
}

// C/D 16x16 f32: VGPR r -> M = r (lanes 0-15) or r+8 (lanes 16-31); N = lane&15.
__device__ __forceinline__ void store_tile_bf16_bias(__bf16* C, int ldc, int m0, int n0,
                                                     v8f acc, const float* bias) {
  int lane = threadIdx.x & 31;
  int col = n0 + (lane & 15);
  int rbase = m0 + ((lane >> 4) << 3);
  float bv = bias[col];
#pragma unroll
  for (int r = 0; r < 8; ++r)
    C[(size_t)(rbase + r) * ldc + col] = to_bf16(acc[r] + bv);
}

__device__ __forceinline__ void store_tile_f32_bias(float* C, int ldc, int m0, int n0,
                                                    v8f acc, const float* bias) {
  int lane = threadIdx.x & 31;
  int col = n0 + (lane & 15);
  int rbase = m0 + ((lane >> 4) << 3);
  float bv = bias[col];
#pragma unroll
  for (int r = 0; r < 8; ++r)
    C[(size_t)(rbase + r) * ldc + col] = acc[r] + bv;
}

// ---------------------------------------------------------------------------
// K-1: one-time f32 -> bf16 conversion (vectorized), n4 = n/4
// ---------------------------------------------------------------------------
__global__ __launch_bounds__(256) void cvt_f32_bf16_kernel(const float4* __restrict__ in,
                                                           ushort4* __restrict__ out, int n4) {
  int i = blockIdx.x * 256 + threadIdx.x;
  if (i < n4) {
    float4 v = in[i];
    ushort4 o;
    o.x = bf16_bits(v.x);
    o.y = bf16_bits(v.y);
    o.z = bf16_bits(v.z);
    o.w = bf16_bits(v.w);
    out[i] = o;
  }
}

// ---------------------------------------------------------------------------
// K0: zero the attention-weight column-sum accumulator
// ---------------------------------------------------------------------------
__global__ void zero_kernel(float* p, int n) {
  int i = blockIdx.x * blockDim.x + threadIdx.x;
  if (i < n) p[i] = 0.0f;
}

// ---------------------------------------------------------------------------
// K1: qkv = x @ w_in^T + b_in   (M=8192, N=3072, K=1024), bf16 in, bf16 out
// 8 waves per WG (2x4); wave tile 32x64 (2 A-frags x 4 B-frags -> 8 WMMA/kstep)
// WG tile 64x256.
// ---------------------------------------------------------------------------
__global__ __launch_bounds__(256) void qkv_proj_kernel(const __bf16* __restrict__ X,
                                                       const __bf16* __restrict__ W,
                                                       const float* __restrict__ bias,
                                                       __bf16* __restrict__ C) {
  const int K = E_DIM, N = THREE_E;
  int wv = threadIdx.x >> 5;
  int wm = wv >> 2, wn = wv & 3;
  int m0 = blockIdx.y * 64 + wm * 32;
  int n0 = blockIdx.x * 256 + wn * 64;
  v8f acc[2][4] = {};
  for (int kk = 0; kk < K; kk += 32) {
    v16bf a0 = frag_a_bf16(X, K, m0, kk);
    v16bf a1 = frag_a_bf16(X, K, m0 + 16, kk);
#pragma unroll
    for (int j = 0; j < 4; ++j) {
      v16bf b = frag_b_nt_bf16(W, K, n0 + j * 16, kk);
      acc[0][j] = wmma_bf16(a0, b, acc[0][j]);
      acc[1][j] = wmma_bf16(a1, b, acc[1][j]);
    }
  }
#pragma unroll
  for (int i = 0; i < 2; ++i)
#pragma unroll
    for (int j = 0; j < 4; ++j)
      store_tile_bf16_bias(C, N, m0 + i * 16, n0 + j * 16, acc[i][j], bias);
}

// ---------------------------------------------------------------------------
// K2: fused attention for one (b, h, 16-row s-block); 1 wave per block.
//   scores = scale * Q K^T  -> LDS (16x1024 f32, 64 KB)
//   softmax rows in LDS; column sums atomically accumulated into cw[b, t]
//   ctx = P @ V (16x64), stored bf16
// ---------------------------------------------------------------------------
__global__ __launch_bounds__(32) void attn_kernel(const __bf16* __restrict__ qkv,
                                                  float* __restrict__ cw,
                                                  __bf16* __restrict__ ctx) {
  __shared__ float sm[16 * S_DIM];   // 64 KB
  const int ldr = B_DIM * THREE_E;   // stride between consecutive s/t rows
  int lane = threadIdx.x;
  int s0 = blockIdx.x * 16;
  int h  = blockIdx.y;
  int b  = blockIdx.z;
  const __bf16* qb = qkv + (size_t)b * THREE_E + h * D_DIM;
  const __bf16* kb = qb + E_DIM;
  const __bf16* vb = qb + 2 * E_DIM;
  const float scale = 0.125f;  // 1/sqrt(64)

  // ---- scores: Q (hoisted) x K^T, 64 tiles along t ----
  v16bf qa0 = frag_a_bf16(qb, ldr, s0, 0);
  v16bf qa1 = frag_a_bf16(qb, ldr, s0, 32);
  for (int t0 = 0; t0 < S_DIM; t0 += 16) {
    v8f acc = {};
    acc = wmma_bf16(qa0, frag_b_nt_bf16(kb, ldr, t0, 0), acc);
    acc = wmma_bf16(qa1, frag_b_nt_bf16(kb, ldr, t0, 32), acc);
    int col = t0 + (lane & 15);
    int rbase = (lane >> 4) << 3;
#pragma unroll
    for (int r = 0; r < 8; ++r) sm[(rbase + r) * S_DIM + col] = acc[r] * scale;
  }
  __syncthreads();

  // ---- softmax: lanes l and l+16 cooperate on row l&15 ----
  {
    int row = lane & 15;
    int half = lane >> 4;
    float* rp = &sm[row * S_DIM + half * 512];
    float mx = -3.4e38f;
    for (int c = 0; c < 512; ++c) mx = fmaxf(mx, rp[c]);
    mx = fmaxf(mx, __shfl_xor(mx, 16, 32));
    float sum = 0.0f;
    for (int c = 0; c < 512; ++c) {
      float e = __expf(rp[c] - mx);
      rp[c] = e;
      sum += e;
    }
    sum += __shfl_xor(sum, 16, 32);
    float inv = 1.0f / sum;
    for (int c = 0; c < 512; ++c) rp[c] *= inv;
  }
  __syncthreads();

  // ---- column sums of attn probs -> cw[b, t] (summed over s and h) ----
  for (int c = lane; c < S_DIM; c += 32) {
    float csum = 0.0f;
#pragma unroll
    for (int r = 0; r < 16; ++r) csum += sm[r * S_DIM + c];
    atomicAdd(&cw[b * S_DIM + c], csum);
  }

  // ---- ctx = P @ V : 16x64 (4 N-tiles), K over t ----
  v8f accv[4] = {};
  for (int kk = 0; kk < S_DIM; kk += 32) {
    v16bf a = frag_a_f32(sm, S_DIM, 0, kk);  // probs from LDS -> bf16
#pragma unroll
    for (int j = 0; j < 4; ++j)
      accv[j] = wmma_bf16(a, frag_b_nn_bf16(vb, ldr, kk, j * 16), accv[j]);
  }
  int colb = lane & 15;
  int rb = (lane >> 4) << 3;
#pragma unroll
  for (int j = 0; j < 4; ++j) {
#pragma unroll
    for (int r = 0; r < 8; ++r) {
      int srow = s0 + rb + r;
      int col = h * D_DIM + j * 16 + colb;
      ctx[((size_t)srow * B_DIM + b) * E_DIM + col] = to_bf16(accv[j][r]);
    }
  }
}

// ---------------------------------------------------------------------------
// K3: attn_out = ctx @ w_out^T + b_out  (M=8192, N=1024, K=1024), f32 out
// ---------------------------------------------------------------------------
__global__ __launch_bounds__(256) void out_proj_kernel(const __bf16* __restrict__ A,
                                                       const __bf16* __restrict__ W,
                                                       const float* __restrict__ bias,
                                                       float* __restrict__ C) {
  const int K = E_DIM, N = E_DIM;
  int wv = threadIdx.x >> 5;
  int wm = wv >> 2, wn = wv & 3;
  int m0 = blockIdx.y * 64 + wm * 32;
  int n0 = blockIdx.x * 256 + wn * 64;
  v8f acc[2][4] = {};
  for (int kk = 0; kk < K; kk += 32) {
    v16bf a0 = frag_a_bf16(A, K, m0, kk);
    v16bf a1 = frag_a_bf16(A, K, m0 + 16, kk);
#pragma unroll
    for (int j = 0; j < 4; ++j) {
      v16bf b = frag_b_nt_bf16(W, K, n0 + j * 16, kk);
      acc[0][j] = wmma_bf16(a0, b, acc[0][j]);
      acc[1][j] = wmma_bf16(a1, b, acc[1][j]);
    }
  }
#pragma unroll
  for (int i = 0; i < 2; ++i)
#pragma unroll
    for (int j = 0; j < 4; ++j)
      store_tile_f32_bias(C, N, m0 + i * 16, n0 + j * 16, acc[i][j], bias);
}

// ---------------------------------------------------------------------------
// K4: pooled[b,e] = (1/(S*H)) * sum_t cw[b,t] * attn_out[(t*B+b)*E + e]
// ---------------------------------------------------------------------------
__global__ __launch_bounds__(256) void pool_kernel(const float* __restrict__ cw,
                                                   const float* __restrict__ ao,
                                                   float* __restrict__ out) {
  int e = blockIdx.x * 256 + threadIdx.x;
  int b = blockIdx.y;
  float acc = 0.0f;
  for (int t = 0; t < S_DIM; ++t)
    acc += cw[b * S_DIM + t] * ao[((size_t)t * B_DIM + b) * E_DIM + e];
  out[b * E_DIM + e] = acc * (1.0f / (float)(S_DIM * H_DIM));
}

// ---------------------------------------------------------------------------
// launch
// ---------------------------------------------------------------------------
extern "C" void kernel_launch(void* const* d_in, const int* in_sizes, int n_in,
                              void* d_out, int out_size, void* d_ws, size_t ws_size,
                              hipStream_t stream) {
  const float* x     = (const float*)d_in[0];
  const float* w_in  = (const float*)d_in[1];
  const float* b_in  = (const float*)d_in[2];
  const float* w_out = (const float*)d_in[3];
  const float* b_out = (const float*)d_in[4];
  float* out = (float*)d_out;

  // workspace layout (bytes):
  //   qkv  bf16 : 8192*3072*2 = 50331648   @ 0
  //   ctx  bf16 : 8192*1024*2 = 16777216   @ 50331648
  //   aout f32  : 8192*1024*4 = 33554432   @ 67108864
  //   cw   f32  : 8*1024*4    = 32768      @ 100663296
  //   x_bf16    : 8192*1024*2 = 16777216   @ 100696064
  //   win_bf16  : 3072*1024*2 = 6291456    @ 117473280
  //   wout_bf16 : 1024*1024*2 = 2097152    @ 123764736
  char* ws = (char*)d_ws;
  __bf16* qkv   = (__bf16*)ws;
  __bf16* ctx   = (__bf16*)(ws + 50331648u);
  float*  aout  = (float*)(ws + 67108864u);
  float*  cw    = (float*)(ws + 100663296u);
  __bf16* xb    = (__bf16*)(ws + 100696064u);
  __bf16* winb  = (__bf16*)(ws + 117473280u);
  __bf16* woutb = (__bf16*)(ws + 123764736u);

  // one-time f32 -> bf16 conversions (bandwidth-bound, done once)
  {
    int n4x = (M_DIM * E_DIM) / 4;       // 2097152
    int n4wi = (THREE_E * E_DIM) / 4;    // 786432
    int n4wo = (E_DIM * E_DIM) / 4;      // 262144
    cvt_f32_bf16_kernel<<<(n4x + 255) / 256, 256, 0, stream>>>((const float4*)x, (ushort4*)xb, n4x);
    cvt_f32_bf16_kernel<<<(n4wi + 255) / 256, 256, 0, stream>>>((const float4*)w_in, (ushort4*)winb, n4wi);
    cvt_f32_bf16_kernel<<<(n4wo + 255) / 256, 256, 0, stream>>>((const float4*)w_out, (ushort4*)woutb, n4wo);
  }
  zero_kernel<<<(B_DIM * S_DIM + 255) / 256, 256, 0, stream>>>(cw, B_DIM * S_DIM);

  qkv_proj_kernel<<<dim3(THREE_E / 256, M_DIM / 64), 256, 0, stream>>>(xb, winb, b_in, qkv);
  attn_kernel<<<dim3(S_DIM / 16, H_DIM, B_DIM), 32, 0, stream>>>(qkv, cw, ctx);
  out_proj_kernel<<<dim3(E_DIM / 256, M_DIM / 64), 256, 0, stream>>>(ctx, woutb, b_out, aout);
  pool_kernel<<<dim3(E_DIM / 256, B_DIM), 256, 0, stream>>>(cw, aout, out);
}
